// NT_Xent_16338055594021
// MI455X (gfx1250) — matrix operand order repaired and checked by reading, not compiled, not run on hardware
//
#include <hip/hip_runtime.h>
#include <hip/hip_bf16.h>
#include <math.h>

#define NROWS 4096   // N = 2*B
#define BATCH 2048
#define DIM   256
#define NCLS  100
#define COLSPLIT 8   // column splits -> 32 rowBlocks * 8 = 256 blocks
#define PITCH 264    // LDS row pitch (elems): 528B -> lanes hit banks 4*l15 (conflict-free)

typedef __attribute__((ext_vector_type(8)))  __bf16 v8bf;
typedef __attribute__((ext_vector_type(16))) __bf16 v16bf;
typedef __attribute__((ext_vector_type(8)))  float  v8f;

// Pointer types matching the async-to-LDS builtin's prototype
// (diagnostic showed: parameter is `int __vector_size(16) __device__ *`).
typedef int v4i __attribute__((vector_size(16)));
typedef __attribute__((address_space(1))) v4i* as1_v4i_p;
typedef __attribute__((address_space(3))) v4i* as3_v4i_p;

union V16U { v16bf v; v8bf h[2]; };

#if __has_builtin(__builtin_amdgcn_global_load_async_to_lds_b128)
#define ASYNC_PATH 1
#else
#define ASYNC_PATH 0
#endif

// 16-byte global -> LDS copy. Async (ASYNCcnt-tracked DMA) when available.
__device__ __forceinline__ void cp16_g2l(const __bf16* g, __bf16* l) {
#if ASYNC_PATH
  __builtin_amdgcn_global_load_async_to_lds_b128(
      (as1_v4i_p)(uintptr_t)g,
      (as3_v4i_p)(uint32_t)(uintptr_t)l,
      0, 0);
#else
  *(v8bf*)l = *(const v8bf*)g;
#endif
}

template <int N>
__device__ __forceinline__ void wait_async() {
#if ASYNC_PATH
#if __has_builtin(__builtin_amdgcn_s_wait_asynccnt)
  __builtin_amdgcn_s_wait_asynccnt(N);
#else
  asm volatile("s_wait_asynccnt %0" :: "i"(N) : "memory");
#endif
#endif
}

// ---------------------------------------------------------------------------
// Kernel 1: row L2-normalize of concat(z_i, z_j) and split into bf16 hi+lo.
// ---------------------------------------------------------------------------
__global__ __launch_bounds__(256) void prep_kernel(const float* __restrict__ z_i,
                                                   const float* __restrict__ z_j,
                                                   __bf16* __restrict__ Zhi,
                                                   __bf16* __restrict__ Zlo) {
  int n = blockIdx.x;
  int d = threadIdx.x;
  float x = (n < BATCH) ? z_i[n * DIM + d] : z_j[(n - BATCH) * DIM + d];
  __shared__ float s[256];
  s[d] = x * x;
  __syncthreads();
  for (int st = 128; st > 0; st >>= 1) {
    if (d < st) s[d] += s[d + st];
    __syncthreads();
  }
  float zn = x * (1.0f / sqrtf(s[0]));
  __bf16 hi = (__bf16)zn;
  __bf16 lo = (__bf16)(zn - (float)hi);
  Zhi[n * DIM + d] = hi;
  Zlo[n * DIM + d] = lo;
}

// ---------------------------------------------------------------------------
// Kernel 2: recover integer labels from one-hot rows (dist@dist^T == label-eq).
// ---------------------------------------------------------------------------
__global__ __launch_bounds__(256) void label_kernel(const float* __restrict__ dist,
                                                    int* __restrict__ labels) {
  int i = blockIdx.x * 256 + threadIdx.x;
  if (i >= BATCH) return;
  int lab = 0;
  for (int c = 0; c < NCLS; ++c)
    if (dist[i * NCLS + c] > 0.5f) lab = c;
  labels[i]         = lab;
  labels[i + BATCH] = lab;
}

// ---------------------------------------------------------------------------
// Kernel 3: fused Gram-matrix (bf16x3 WMMA) + masked row reductions.
// Block = 8 waves = 128 rows x one column split; B tiles staged in LDS
// (double-buffered, async DMA) and shared by all 8 waves -> 8x less L2 traffic.
// ---------------------------------------------------------------------------
__global__ __launch_bounds__(256) void sim_kernel(const __bf16* __restrict__ Zhi,
                                                  const __bf16* __restrict__ Zlo,
                                                  const int* __restrict__ labels,
                                                  float* __restrict__ nomP,
                                                  float* __restrict__ denP) {
  __shared__ __bf16 Bt[2][2][16 * PITCH];   // [buf][hi/lo][row*PITCH + k]

  int tid  = threadIdx.x;
  int w    = tid >> 5;
  int lane = tid & 31;
  int half = lane >> 4;
  int l15  = lane & 15;
  int rowBlock = blockIdx.x >> 3;           // 0..31
  int split    = blockIdx.x & (COLSPLIT - 1);
  int rowBase  = rowBlock * 128 + w * 16;

  // ---- Preload A fragments (16-row strip, full K=256), hi & lo. ----
  // A-layout (16-bit 16x32): elems 0-7 = K(half*8+0..7), elems 8-15 = K(+16).
  int rA = rowBase + l15;
  v16bf ahi[8], alo[8];
#pragma unroll
  for (int ks = 0; ks < 8; ++ks) {
    const __bf16* pa = Zhi + rA * DIM + ks * 32 + half * 8;
    V16U u; u.h[0] = *(const v8bf*)pa; u.h[1] = *(const v8bf*)(pa + 16);
    ahi[ks] = u.v;
    const __bf16* pl = Zlo + rA * DIM + ks * 32 + half * 8;
    V16U t; t.h[0] = *(const v8bf*)pl; t.h[1] = *(const v8bf*)(pl + 16);
    alo[ks] = t.v;
  }

  int labR[8];
#pragma unroll
  for (int vg = 0; vg < 8; ++vg) labR[vg] = labels[rowBase + vg + half * 8];

  float nomAcc[8], denAcc[8];
#pragma unroll
  for (int vg = 0; vg < 8; ++vg) { nomAcc[vg] = 0.f; denAcc[vg] = 0.f; }

  const int ntiles = (NROWS / 16) / COLSPLIT;   // 32 tiles per block

  // Stage one 16x256 (hi+lo) B tile: 1024 16B chunks, 4 per thread.
  auto issue_tile = [&](int ti, int buf) {
    int colBase = (split + ti * COLSPLIT) * 16;
#pragma unroll
    for (int j = 0; j < 4; ++j) {
      int c   = tid + j * 256;          // 0..1023
      int hl  = c >> 9;                 // 0=hi, 1=lo
      int rem = c & 511;
      int row = rem >> 5;               // 0..15
      int k16 = rem & 31;               // 16B chunk within row
      const __bf16* g = (hl ? Zlo : Zhi) + (colBase + row) * DIM + k16 * 8;
      __bf16* l = &Bt[buf][hl][row * PITCH + k16 * 8];
      cp16_g2l(g, l);
    }
  };

  issue_tile(0, 0);

  for (int i = 0; i < ntiles; ++i) {
    int cur = i & 1;
    if (i + 1 < ntiles) {
      issue_tile(i + 1, cur ^ 1);
      wait_async<4>();                  // in-order per wave: tile i complete
    } else {
      wait_async<0>();
    }
    __syncthreads();                    // B tile visible to all 8 waves

    int colBase = (split + i * COLSPLIT) * 16;
    int cB = colBase + l15;             // this lane's output column
    int labC = labels[cB];
    const __bf16* bH = &Bt[cur][0][l15 * PITCH + half * 8];
    const __bf16* bL = &Bt[cur][1][l15 * PITCH + half * 8];

    v8f acc = {};
#pragma unroll
    for (int ks = 0; ks < 8; ++ks) {
      // B-layout (16-bit 32x16): elems 0-7 = K(half*8+0..7), elems 8-15 = K(+16), col = l15.
      V16U bh; bh.h[0] = *(const v8bf*)(bH + ks * 32);
               bh.h[1] = *(const v8bf*)(bH + ks * 32 + 16);
      V16U bl; bl.h[0] = *(const v8bf*)(bL + ks * 32);
               bl.h[1] = *(const v8bf*)(bL + ks * 32 + 16);
      // sim = hi*hi + hi*lo + lo*hi  (lo*lo ~ 2^-18, negligible)
      acc = __builtin_amdgcn_wmma_f32_16x16x32_bf16(false, ahi[ks], false, bh.v,
                                                    (short)0, acc, false, false);
      acc = __builtin_amdgcn_wmma_f32_16x16x32_bf16(false, ahi[ks], false, bl.v,
                                                    (short)0, acc, false, false);
      acc = __builtin_amdgcn_wmma_f32_16x16x32_bf16(false, alo[ks], false, bh.v,
                                                    (short)0, acc, false, false);
    }

    // C/D layout: VGPR vg, lanes 0-15 -> row vg, lanes 16-31 -> row vg+8.
#pragma unroll
    for (int vg = 0; vg < 8; ++vg) {
      int r = rowBase + vg + half * 8;
      float s = acc[vg];
      bool offd = (r != cB);
      if (offd && labR[vg] == labC) nomAcc[vg] += s;
      if (offd)                     denAcc[vg] += __expf(s);
    }
    __syncthreads();                    // done reading buf before it's re-filled
  }

  // Reduce each row across its 16 lanes (xor masks < 16 stay within halves).
#pragma unroll
  for (int vg = 0; vg < 8; ++vg) {
    float nv = nomAcc[vg], dv = denAcc[vg];
#pragma unroll
    for (int off = 8; off > 0; off >>= 1) {
      nv += __shfl_xor(nv, off, 32);
      dv += __shfl_xor(dv, off, 32);
    }
    if (l15 == 0) {
      int r = rowBase + vg + half * 8;
      nomP[split * NROWS + r] = nv;
      denP[split * NROWS + r] = dv;
    }
  }
}

// ---------------------------------------------------------------------------
// Kernel 4: combine column-split partials, loss = sum(nom/den) / N.
// ---------------------------------------------------------------------------
__global__ __launch_bounds__(256) void loss_kernel(const float* __restrict__ nomP,
                                                   const float* __restrict__ denP,
                                                   float* __restrict__ out) {
  __shared__ float s[256];
  float acc = 0.f;
  for (int r = threadIdx.x; r < NROWS; r += 256) {
    float nv = 0.f, dv = 0.f;
#pragma unroll
    for (int p = 0; p < COLSPLIT; ++p) {
      nv += nomP[p * NROWS + r];
      dv += denP[p * NROWS + r];
    }
    acc += nv / dv;
  }
  s[threadIdx.x] = acc;
  __syncthreads();
  for (int st = 128; st > 0; st >>= 1) {
    if (threadIdx.x < st) s[threadIdx.x] += s[threadIdx.x + st];
    __syncthreads();
  }
  if (threadIdx.x == 0) out[0] = s[0] / (float)NROWS;
}

// ---------------------------------------------------------------------------
extern "C" void kernel_launch(void* const* d_in, const int* in_sizes, int n_in,
                              void* d_out, int out_size, void* d_ws, size_t ws_size,
                              hipStream_t stream) {
  const float* z_i  = (const float*)d_in[0];
  const float* z_j  = (const float*)d_in[1];
  // d_in[2] (z_n) is unused by the reference loss
  const float* dist = (const float*)d_in[3];

  char* ws = (char*)d_ws;
  __bf16* Zhi    = (__bf16*)(ws);                                   // 2 MB
  __bf16* Zlo    = (__bf16*)(ws + (size_t)2 * 1024 * 1024);         // 2 MB
  int*    labels = (int*)  (ws + (size_t)4 * 1024 * 1024);          // 16 KB
  float*  nomP   = (float*)(ws + (size_t)4 * 1024 * 1024 + 64 * 1024);              // 128 KB
  float*  denP   = (float*)(ws + (size_t)4 * 1024 * 1024 + 64 * 1024 + 128 * 1024); // 128 KB

  prep_kernel <<<NROWS, 256, 0, stream>>>(z_i, z_j, Zhi, Zlo);
  label_kernel<<<(BATCH + 255) / 256, 256, 0, stream>>>(dist, labels);
  sim_kernel  <<<(NROWS / 128) * COLSPLIT, 256, 0, stream>>>(Zhi, Zlo, labels, nomP, denP);
  loss_kernel <<<1, 256, 0, stream>>>(nomP, denP, (float*)d_out);
}
